// LSTM_9457517985995
// MI455X (gfx1250) — matrix operand order repaired and checked by reading, not compiled, not run on hardware
//
#include <hip/hip_runtime.h>
#include <hip/hip_bf16.h>

// Problem constants (match reference)
#define Bdim 4
#define Tdim 1024
#define Hdim 512
#define Odim 32000
#define G4H  (4 * Hdim)      // 2048 gate rows per layer
#define Mdim (Tdim * Bdim)   // 4096 flattened (t,b) rows
#define BH   (Bdim * Hdim)   // 2048
#define NWG  64              // blocks in persistent scan kernel
#define UPW  (Hdim / NWG)    // hidden units per scan block = 8

typedef __attribute__((ext_vector_type(16))) _Float16 v16h;
typedef __attribute__((ext_vector_type(8)))  _Float16 v8h;
typedef __attribute__((ext_vector_type(8)))  float    v8f;
typedef __attribute__((ext_vector_type(4)))  int      v4i;

#if defined(__has_builtin)
#if __has_builtin(__builtin_amdgcn_global_load_async_to_lds_b128) && \
    __has_builtin(__builtin_amdgcn_s_wait_asynccnt)
#define HAVE_ASYNC_LDS 1
#endif
#endif
#ifndef HAVE_ASYNC_LDS
#define HAVE_ASYNC_LDS 0
#endif

#if HAVE_ASYNC_LDS
typedef __attribute__((address_space(1))) v4i* gbl_v4i_ptr;
typedef __attribute__((address_space(3))) v4i* lds_v4i_ptr;
#endif

// 16-byte global -> LDS copy; async (ASYNCcnt-tracked) when available.
__device__ __forceinline__ void copy_b128_to_lds(const _Float16* gsrc, _Float16* ldst) {
#if HAVE_ASYNC_LDS
    __builtin_amdgcn_global_load_async_to_lds_b128(
        (gbl_v4i_ptr)(_Float16*)gsrc,
        (lds_v4i_ptr)ldst,
        0, 0);
#else
    *(v8h*)ldst = *(const v8h*)gsrc;
#endif
}

__device__ __forceinline__ void wait_async_copies() {
#if HAVE_ASYNC_LDS
    __builtin_amdgcn_s_wait_asynccnt(0);
#endif
}

// ---------------------------------------------------------------------------
// Zero-init (workspace is poisoned with 0xAA by harness)
// ---------------------------------------------------------------------------
__global__ void zero_ws(unsigned* __restrict__ p, int n) {
    int i = blockIdx.x * 256 + threadIdx.x;
    if (i < n) p[i] = 0u;
}

// ---------------------------------------------------------------------------
// f32 -> f16 conversion
// ---------------------------------------------------------------------------
__global__ void f32_to_f16(const float* __restrict__ in, _Float16* __restrict__ out, int n) {
    int i = blockIdx.x * 256 + threadIdx.x;
    if (i < n) out[i] = (_Float16)in[i];
}

// ---------------------------------------------------------------------------
// Embedding gather: x[t*B*H + b*H + h] = embed[tokens[b*T + t]][h]  (f16)
// ---------------------------------------------------------------------------
__global__ void embed_gather(const int* __restrict__ tok, const float* __restrict__ embed,
                             _Float16* __restrict__ x, int total) {
    int i = blockIdx.x * 256 + threadIdx.x;
    if (i >= total) return;
    int h  = i & (Hdim - 1);
    int bt = i >> 9;            // t*B + b
    int b  = bt & 3;
    int t  = bt >> 2;
    int token = tok[b * Tdim + t];
    x[i] = (_Float16)embed[(size_t)token * Hdim + h];
}

// ---------------------------------------------------------------------------
// WMMA GEMM: C[m,n] = sum_k A[m,k] * W[n,k] + bias[n]
//   A: [M,K] f16 row-major, W: [N,K] f16 row-major (B = W^T), C: f32.
//   Block tile: 128 M-rows (wave w -> M-tile w) x 64 N-cols shared by all waves.
//   Per 32-k step, a 64x32 f16 slab of W is staged to LDS (double-buffered,
//   async-to-LDS when available) and consumed by all 8 waves -> 8x W reuse.
//   out_mode 0: C row-major [M,N]
//   out_mode 1: m = t*B + b  ->  C[b*T*O + t*O + n]   (final logits layout)
// ---------------------------------------------------------------------------
__global__ __launch_bounds__(256)
void gemm_wmma_f16(const _Float16* __restrict__ A,
                   const _Float16* __restrict__ W,
                   const float* __restrict__ bias,
                   float* __restrict__ C,
                   int M, int N, int K, int out_mode) {
    __shared__ _Float16 bsm[2][64 * 32];      // two 4KB W slabs

    const int tid  = threadIdx.x;
    const int lane = tid & 31;
    const int wave = tid >> 5;
    const int m0   = blockIdx.x * 128 + wave * 16;
    const int n0   = blockIdx.y * 64;         // N is a multiple of 64
    const int r    = lane & 15;
    const int half = lane >> 4;

    // A fragment source (16-bit A 16x32): lane holds row (lane&15),
    // contiguous K [half*16 .. +15] (matches B fragment order)
    const _Float16* arow = A + (size_t)(m0 + r) * K;

    // staging: thread tid copies 16B -> slab[row=tid>>2][colh=(tid&3)*8]
    const int crow = tid >> 2;
    const int ccol = (tid & 3) * 8;
    const _Float16* gsrc = W + (size_t)(n0 + crow) * K + ccol;

    // prologue: fill slab 0 with k-slab [0,32)
    copy_b128_to_lds(gsrc, &bsm[0][tid * 8]);

    v8f acc[4] = {};
    int p = 0;
    for (int k0 = 0; k0 < K; k0 += 32, p ^= 1) {
        wait_async_copies();
        __syncthreads();                       // slab p fully resident
        if (k0 + 32 < K)                       // prefetch next slab
            copy_b128_to_lds(gsrc + k0 + 32, &bsm[p ^ 1][tid * 8]);

        v8h alo = *(const v8h*)(arow + k0 + half * 16);
        v8h ahi = *(const v8h*)(arow + k0 + half * 16 + 8);
        v16h a = __builtin_shufflevector(alo, ahi,
                 0,1,2,3,4,5,6,7,8,9,10,11,12,13,14,15);

        #pragma unroll
        for (int j = 0; j < 4; ++j) {
            // B fragment (32x16): lane holds col (lane&15), contiguous K [half*16 .. +15]
            v16h b = *(const v16h*)&bsm[p][(j * 16 + r) * 32 + half * 16];
            acc[j] = __builtin_amdgcn_wmma_f32_16x16x32_f16(
                false, a, false, b, (short)0, acc[j], false, false);
        }
        // slab p is next overwritten only by the copy issued after the barrier
        // at the top of iteration k0+32 -> no trailing barrier needed.
    }

    // C layout (ISA 7.12.2): lane holds col (lane&15); VGPR v holds row v + 8*half
    #pragma unroll
    for (int j = 0; j < 4; ++j) {
        int col = n0 + j * 16 + r;
        float bv = bias ? bias[col] : 0.0f;
        #pragma unroll
        for (int v = 0; v < 8; ++v) {
            int m = m0 + half * 8 + v;
            float val = acc[j][v] + bv;
            if (out_mode == 0) {
                C[(size_t)m * N + col] = val;
            } else {
                int b = m & 3, t = m >> 2;
                C[((size_t)b * Tdim + t) * Odim + col] = val;
            }
        }
    }
}

// ---------------------------------------------------------------------------
// Persistent LSTM scan for one layer. G = precomputed x@W_ih^T + b  [T,B,4H].
// 64 blocks x 256 threads; each block owns 8 hidden units (4 gate rows each).
// Grid-wide barrier per timestep via monotonic atomic counter; h double-buffered.
// Recurrence kept in f32 (error does not compound through f16).
// ---------------------------------------------------------------------------
__global__ __launch_bounds__(256)
void lstm_scan(const float* __restrict__ G,       // [T, B, 4H]
               const float* __restrict__ Whh,     // [4H, H] (this layer)
               float* __restrict__ hbuf,          // [2, B, H] zeroed
               float* __restrict__ cbuf,          // [B, H] zeroed
               _Float16* __restrict__ hseq,       // [T, B, H] f16 out
               unsigned* __restrict__ ctr) {      // zeroed
    __shared__ float hsm[BH];     // staged h_prev (8 KB)
    __shared__ float gsm[128];    // dot-product results

    const int tid  = threadIdx.x;
    const int d    = tid >> 1;            // dot index 0..127
    const int part = tid & 1;             // half of K
    const int u    = d >> 4;              // unit within block 0..7
    const int g    = (d >> 2) & 3;        // gate i/f/g/o
    const int b    = d & 3;               // batch
    const int j    = blockIdx.x * UPW + u;        // global hidden unit
    const int row  = g * Hdim + j;                // gate row
    const float* wrow = Whh + (size_t)row * Hdim + part * (Hdim / 2);

    for (int t = 0; t < Tdim; ++t) {
        // stage h(t-1) into LDS
        const float* hp = hbuf + (t & 1) * BH;
        for (int i = tid; i < BH; i += 256) hsm[i] = hp[i];
        __syncthreads();

        // partial dot: 256 MACs per thread
        const float* hrow = hsm + b * Hdim + part * (Hdim / 2);
        float s = 0.0f;
        #pragma unroll 8
        for (int k = 0; k < Hdim / 2; ++k) s += wrow[k] * hrow[k];
        s += __shfl_xor(s, 1, 32);
        if (part == 0) gsm[d] = s;
        __syncthreads();

        // gate nonlinearities + state update: 32 threads (8 units x 4 batch)
        if (tid < 32) {
            int uu = tid >> 2, bb = tid & 3;
            int jj = blockIdx.x * UPW + uu;
            int base = uu * 16 + bb;
            const float* Gt = G + ((size_t)t * Bdim + bb) * G4H;
            float gi = gsm[base + 0]  + Gt[0 * Hdim + jj];
            float gf = gsm[base + 4]  + Gt[1 * Hdim + jj];
            float gg = gsm[base + 8]  + Gt[2 * Hdim + jj];
            float go = gsm[base + 12] + Gt[3 * Hdim + jj];
            gi = 1.0f / (1.0f + expf(-gi));
            gf = 1.0f / (1.0f + expf(-gf));
            gg = tanhf(gg);
            go = 1.0f / (1.0f + expf(-go));
            float c  = gf * cbuf[bb * Hdim + jj] + gi * gg;
            float hn = go * tanhf(c);
            cbuf[bb * Hdim + jj] = c;
            hbuf[((t + 1) & 1) * BH + bb * Hdim + jj] = hn;
            hseq[((size_t)t * Bdim + bb) * Hdim + jj] = (_Float16)hn;
        }

        // grid-wide barrier (release h(t) writes, acquire everyone else's)
        __threadfence();
        __syncthreads();
        if (tid == 0) {
            atomicAdd(ctr, 1u);
            unsigned target = (unsigned)gridDim.x * (unsigned)(t + 1);
            while (__hip_atomic_load(ctr, __ATOMIC_ACQUIRE, __HIP_MEMORY_SCOPE_AGENT) < target) {
                __builtin_amdgcn_s_sleep(1);
            }
        }
        __syncthreads();
        __threadfence();
    }
}

// ---------------------------------------------------------------------------
// Host-side launch
// ---------------------------------------------------------------------------
extern "C" void kernel_launch(void* const* d_in, const int* in_sizes, int n_in,
                              void* d_out, int out_size, void* d_ws, size_t ws_size,
                              hipStream_t stream) {
    const int*   tokens = (const int*)d_in[0];
    const float* embed  = (const float*)d_in[1];
    const float* W_ih   = (const float*)d_in[2];   // [2, 2048, 512]
    const float* W_hh   = (const float*)d_in[3];   // [2, 2048, 512]
    const float* bvec   = (const float*)d_in[4];   // [2, 2048]
    const float* W_out  = (const float*)d_in[5];   // [32000, 512]
    const float* b_out  = (const float*)d_in[6];   // [32000]
    float* out = (float*)d_out;

    char* ws = (char*)d_ws;
    size_t off = 0;
    auto alloc = [&](size_t bytes) -> void* {
        void* p = ws + off;
        off = (off + bytes + 255) & ~(size_t)255;
        return p;
    };

    float*    hbuf  = (float*)alloc(2 * BH * sizeof(float));
    float*    cbuf  = (float*)alloc(BH * sizeof(float));
    unsigned* ctrs  = (unsigned*)alloc(256);
    size_t hc_bytes = off;                               // region to zero before each scan
    _Float16* x0    = (_Float16*)alloc((size_t)Mdim * Hdim * 2);
    _Float16* h1    = (_Float16*)alloc((size_t)Mdim * Hdim * 2);
    _Float16* h2    = (_Float16*)alloc((size_t)Mdim * Hdim * 2);
    _Float16* wih16 = (_Float16*)alloc((size_t)2 * G4H * Hdim * 2);
    _Float16* wo16  = (_Float16*)alloc((size_t)Odim * Hdim * 2);
    float*    Gbuf  = (float*)alloc((size_t)Mdim * G4H * 4);  // reused for both layers

    const int hc_dwords = (int)(hc_bytes / 4);

    // 1) zero h/c/counters; stage weights in f16; gather embeddings
    zero_ws<<<(hc_dwords + 255) / 256, 256, 0, stream>>>((unsigned*)ws, hc_dwords);
    {
        int n = 2 * G4H * Hdim;
        f32_to_f16<<<(n + 255) / 256, 256, 0, stream>>>(W_ih, wih16, n);
    }
    {
        int n = Odim * Hdim;
        f32_to_f16<<<(n + 255) / 256, 256, 0, stream>>>(W_out, wo16, n);
    }
    {
        int n = Mdim * Hdim;
        embed_gather<<<(n + 255) / 256, 256, 0, stream>>>(tokens, embed, x0, n);
    }

    // 2) layer 0: G0 = x @ W_ih[0]^T + b[0], then sequential scan
    gemm_wmma_f16<<<dim3(Mdim / 128, G4H / 64), 256, 0, stream>>>(
        x0, wih16, bvec, Gbuf, Mdim, G4H, Hdim, 0);
    lstm_scan<<<NWG, 256, 0, stream>>>(Gbuf, W_hh, hbuf, cbuf, h1, ctrs);

    // 3) layer 1: G1 = h1 @ W_ih[1]^T + b[1], re-zero state, scan
    gemm_wmma_f16<<<dim3(Mdim / 128, G4H / 64), 256, 0, stream>>>(
        h1, wih16 + (size_t)G4H * Hdim, bvec + G4H, Gbuf, Mdim, G4H, Hdim, 0);
    zero_ws<<<(hc_dwords + 255) / 256, 256, 0, stream>>>((unsigned*)ws, hc_dwords);
    lstm_scan<<<NWG, 256, 0, stream>>>(Gbuf, W_hh + (size_t)G4H * Hdim, hbuf, cbuf, h2, ctrs);

    // 4) logits: out[b,t,:] = h2[t,b,:] @ W_out^T + b_out   (134 GFLOP, WMMA)
    gemm_wmma_f16<<<dim3(Mdim / 128, Odim / 64), 256, 0, stream>>>(
        h2, wo16, b_out, out, Mdim, Odim, Hdim, 1);
}
